// CollaborativeAttention_2121713844252
// MI455X (gfx1250) — compile-verified, matrix operand order checked by reading, not run on hardware
//
#include <hip/hip_runtime.h>
#include <hip/hip_bf16.h>

// ---------------------------------------------------------------------------
// Collaborative (complex) attention for MI455X / gfx1250, wave32 + WMMA bf16.
// B=2, S=1024, D=DK=DV=768, H=12, head_dim=64, scale = sqrt(DK/H) = 8.
// Attention kernel double-buffers K-side tiles via async global->LDS DMA:
// issue next tile's copies, s_wait_asynccnt 26 (in-order retire) to drain the
// current tile, compute overlaps the in-flight DMA.
// ---------------------------------------------------------------------------

typedef __bf16 bf16;
typedef __attribute__((ext_vector_type(16))) __bf16 v16bf;
typedef __attribute__((ext_vector_type(8)))  __bf16 v8bf;
typedef __attribute__((ext_vector_type(8)))  float  v8f;
typedef __attribute__((ext_vector_type(8)))  unsigned int v8u;
typedef __attribute__((ext_vector_type(4)))  unsigned int v4u;

#define B_   2
#define S_   1024
#define D_   768      // D == DK == DV
#define H_   12
#define HD_  64       // DV / H
#define BS_  (B_*S_)  // 2048
#define BH_  (B_*H_)  // 24

struct V8x2 { v8bf lo, hi; };

// A/B fragment load: ISA 7.12.2 16-bit 16x32 layout.
// lane L holds row (L&15); K offsets (L>>4)*8 .. +7 and 16+(L>>4)*8 .. +7.
// -> two contiguous 16-byte loads per fragment.
__device__ __forceinline__ v16bf ldfrag(const bf16* p) {
    V8x2 t;
    t.lo = *reinterpret_cast<const v8bf*>(p);
    t.hi = *reinterpret_cast<const v8bf*>(p + 16);
    return __builtin_bit_cast(v16bf, t);
}

// Fragment with only K=0..15 valid (K=16..31 zeroed) for K=16 tail GEMMs.
__device__ __forceinline__ v16bf ldfrag_lo(const bf16* p) {
    V8x2 t;
    t.lo = *reinterpret_cast<const v8bf*>(p);
    v4u z = {0u, 0u, 0u, 0u};
    t.hi = __builtin_bit_cast(v8bf, z);
    return __builtin_bit_cast(v16bf, t);
}

// Packed bf16 negate (sign-bit flip) of a whole fragment.
__device__ __forceinline__ v16bf negf(v16bf x) {
    v8u u = __builtin_bit_cast(v8u, x);
    u = u ^ 0x80008000u;
    return __builtin_bit_cast(v16bf, u);
}

__device__ __forceinline__ v8f zero8() {
    v8f z = {0.f, 0.f, 0.f, 0.f, 0.f, 0.f, 0.f, 0.f};
    return z;
}

#define WMMA_BF16(A, Bm, C) \
    __builtin_amdgcn_wmma_f32_16x16x32_bf16(false, (A), false, (Bm), (short)0, (C), false, false)

// ---------------------------------------------------------------------------
// gfx1250 async global->LDS copy (16B per lane), tracked by ASYNCcnt.
// ---------------------------------------------------------------------------
__device__ __forceinline__ unsigned lds_off(const void* p) {
    // generic LDS pointer: low 32 bits are the LDS byte offset
    return (unsigned)(uintptr_t)p;
}
__device__ __forceinline__ void async_b128(unsigned lds_dst, const bf16* gsrc) {
    asm volatile("global_load_async_to_lds_b128 %0, %1, off"
                 :: "v"(lds_dst), "v"((unsigned long long)(uintptr_t)gsrc)
                 : "memory");
}
__device__ __forceinline__ void wait_async0() {
    asm volatile("s_wait_asynccnt 0x0" ::: "memory");
}
// Drain all but the most recent 26 async ops (one tile's worth) — in-order
// retirement makes this a "previous tile complete" wait.
__device__ __forceinline__ void wait_async26() {
    asm volatile("s_wait_asynccnt 0x1a" ::: "memory");
}

// ---------------------------------------------------------------------------
// Kernel 1: f32 -> bf16 straight convert
// ---------------------------------------------------------------------------
__global__ void k_cvt(const float* __restrict__ in, bf16* __restrict__ out, int n) {
    int i = blockIdx.x * blockDim.x + threadIdx.x;
    if (i < n) out[i] = (bf16)in[i];
}

// Kernel 2: weight transpose + convert. W is [K=768][N=768]; WT[n*768+k].
__global__ void k_cvtT(const float* __restrict__ W, bf16* __restrict__ WT) {
    int i = blockIdx.x * blockDim.x + threadIdx.x;
    if (i >= D_ * D_) return;
    int n = i / D_, k = i % D_;
    WT[i] = (bf16)W[k * D_ + n];
}

// ---------------------------------------------------------------------------
// Kernel 3: complex projection GEMM  (out_r + i*out_i) = (hr + i*hi) @ (Wr + i*Wi)
// One 16x16 complex output tile per wave; K-loop over 768 in steps of 32.
// blockIdx.z selects q / k / v (v adds bias).
// ---------------------------------------------------------------------------
__global__ __launch_bounds__(128) void k_proj(
    const bf16* __restrict__ hr, const bf16* __restrict__ hi,
    const bf16* __restrict__ WqrT, const bf16* __restrict__ WqiT,
    const bf16* __restrict__ WkrT, const bf16* __restrict__ WkiT,
    const bf16* __restrict__ WvrT, const bf16* __restrict__ WviT,
    const float* __restrict__ bvr, const float* __restrict__ bvi,
    bf16* __restrict__ qr, bf16* __restrict__ qi,
    bf16* __restrict__ kr, bf16* __restrict__ ki,
    bf16* __restrict__ vr, bf16* __restrict__ vi)
{
    int which = blockIdx.z;
    const bf16 *BrT, *BiT;
    bf16 *Or, *Oi;
    if (which == 0)      { BrT = WqrT; BiT = WqiT; Or = qr; Oi = qi; }
    else if (which == 1) { BrT = WkrT; BiT = WkiT; Or = kr; Oi = ki; }
    else                 { BrT = WvrT; BiT = WviT; Or = vr; Oi = vi; }

    int wave = threadIdx.x >> 5, lane = threadIdx.x & 31;
    int l15 = lane & 15, koff = (lane >> 4) * 8;
    int mt = blockIdx.x;              // 0..127  (M tiles, M = 2048)
    int nt = blockIdx.y * 4 + wave;   // 0..47   (N tiles, N = 768)

    const bf16* Ar = hr  + (size_t)(mt * 16 + l15) * D_ + koff;
    const bf16* Ai = hi  + (size_t)(mt * 16 + l15) * D_ + koff;
    const bf16* Br = BrT + (size_t)(nt * 16 + l15) * D_ + koff;
    const bf16* Bi = BiT + (size_t)(nt * 16 + l15) * D_ + koff;

    v8f accr = zero8(), acci = zero8();
    #pragma unroll 4
    for (int k = 0; k < D_; k += 32) {
        v16bf ar  = ldfrag(Ar + k);
        v16bf ai  = ldfrag(Ai + k);
        v16bf br  = ldfrag(Br + k);
        v16bf bi  = ldfrag(Bi + k);
        v16bf ain = negf(ai);
        accr = WMMA_BF16(ar,  br, accr);   // + hr*Wr
        accr = WMMA_BF16(ain, bi, accr);   // - hi*Wi
        acci = WMMA_BF16(ar,  bi, acci);   // + hr*Wi
        acci = WMMA_BF16(ai,  br, acci);   // + hi*Wr
    }

    int n = nt * 16 + l15;
    float biasr = (which == 2) ? bvr[n] : 0.0f;
    float biasi = (which == 2) ? bvi[n] : 0.0f;
    #pragma unroll
    for (int r = 0; r < 8; ++r) {
        int m = mt * 16 + r + (lane >> 4) * 8;   // C-layout row
        Or[(size_t)m * D_ + n] = (bf16)(accr[r] + biasr);
        Oi[(size_t)m * D_ + n] = (bf16)(acci[r] + biasi);
    }
}

// ---------------------------------------------------------------------------
// Kernel 4: content-bias projection (D=768 -> H=12), tiny; keep f32 VALU.
// ---------------------------------------------------------------------------
__global__ void k_cb(const float* __restrict__ hr, const float* __restrict__ hi,
                     const float* __restrict__ Wr, const float* __restrict__ Wi,
                     float* __restrict__ cbr, float* __restrict__ cbi)
{
    int idx = blockIdx.x * blockDim.x + threadIdx.x;   // bs*12 + h
    if (idx >= BS_ * H_) return;
    int h = idx % H_, bs = idx / H_;
    const float* xr = hr + (size_t)bs * D_;
    const float* xi = hi + (size_t)bs * D_;
    float ar = 0.f, ai = 0.f;
    for (int d = 0; d < D_; ++d) {
        float wr = Wr[d * H_ + h], wi = Wi[d * H_ + h];
        float r = xr[d], im = xi[d];
        ar += r * wr - im * wi;
        ai += r * wi + im * wr;
    }
    cbr[idx] = ar;
    cbi[idx] = ai;
}

// ---------------------------------------------------------------------------
// Kernel 5: mixed keys  mk[bh,t,d] = mix[h,d] * k[b,t,d]  (complex, bf16 out)
// ---------------------------------------------------------------------------
__global__ void k_mk(const bf16* __restrict__ kr, const bf16* __restrict__ ki,
                     const float* __restrict__ mixr, const float* __restrict__ mixi,
                     bf16* __restrict__ mkr, bf16* __restrict__ mki)
{
    size_t i = (size_t)blockIdx.x * blockDim.x + threadIdx.x;
    if (i >= (size_t)BH_ * S_ * D_) return;
    int d = (int)(i % D_);
    size_t rest = i / D_;
    int t = (int)(rest % S_);
    int bh = (int)(rest / S_);
    int b = bh / H_, h = bh % H_;
    float krv = (float)kr[(size_t)(b * S_ + t) * D_ + d];
    float kiv = (float)ki[(size_t)(b * S_ + t) * D_ + d];
    float mr = mixr[h * D_ + d], mi = mixi[h * D_ + d];
    mkr[i] = (bf16)(mr * krv - mi * kiv);
    mki[i] = (bf16)(mr * kiv + mi * krv);
}

// ---------------------------------------------------------------------------
// Kernel 6: per-head V transpose: vT[(bh*64+n)*S + t] = v[b,t,h*64+n]
// ---------------------------------------------------------------------------
__global__ void k_vT(const bf16* __restrict__ vr, const bf16* __restrict__ vi,
                     bf16* __restrict__ vTr, bf16* __restrict__ vTi)
{
    int i = blockIdx.x * blockDim.x + threadIdx.x;
    if (i >= BH_ * HD_ * S_) return;
    int t = i % S_;
    int rest = i / S_;
    int n = rest % HD_;
    int bh = rest / HD_;
    int b = bh / H_, h = bh % H_;
    size_t src = (size_t)(b * S_ + t) * D_ + h * HD_ + n;
    vTr[i] = vr[src];
    vTi[i] = vi[src];
}

// ---------------------------------------------------------------------------
// Kernel 7: flash-style complex attention, double-buffered async-LDS staging.
// Block = 128 threads = 4 waves; wave w owns 16 query rows of one (b,h).
// Pipeline per 16-key tile:
//   barrier -> issue next tile's 26 async b128 copies into alternate buffer
//   -> s_wait_asynccnt 26 (drains current tile, in-order) -> barrier
//   -> 96 score WMMAs (K=768) from LDS, dual online softmax, 16 PV WMMAs.
// LDS: 2x(48KB mk + 4KB V^T) + 4KB P-bounce = 108KB (of 320KB per WGP).
// ---------------------------------------------------------------------------
__global__ __launch_bounds__(128) void k_attn(
    const bf16* __restrict__ qr,  const bf16* __restrict__ qi,
    const bf16* __restrict__ mkr, const bf16* __restrict__ mki,
    const bf16* __restrict__ vTr, const bf16* __restrict__ vTi,
    const float* __restrict__ cbr, const float* __restrict__ cbi,
    float* __restrict__ out)
{
    __shared__ alignas(16) bf16 smkr[2][16 * D_];     // 2 x 24 KB
    __shared__ alignas(16) bf16 smki[2][16 * D_];     // 2 x 24 KB
    __shared__ alignas(16) bf16 svr[2][HD_ * 16];     // 2 x 2 KB
    __shared__ alignas(16) bf16 svi[2][HD_ * 16];     // 2 x 2 KB
    __shared__ alignas(16) bf16 pshm[4][2][16 * 16];  // 4 KB (per-wave P tiles)

    int tid  = threadIdx.x;
    int wave = tid >> 5, lane = tid & 31;
    int l15 = lane & 15, lh = lane >> 4;
    int koff = lh * 8;

    int bh = blockIdx.y;
    int b = bh / H_, h = bh % H_;
    int srow = blockIdx.x * 64 + wave * 16;

    const bf16* qrb  = qr + (size_t)(b * S_ + srow + l15) * D_ + koff;
    const bf16* qib  = qi + (size_t)(b * S_ + srow + l15) * D_ + koff;
    const bf16* mkrb = mkr + (size_t)bh * S_ * D_;
    const bf16* mkib = mki + (size_t)bh * S_ * D_;
    const bf16* vTrb = vTr + (size_t)bh * HD_ * S_;
    const bf16* vTib = vTi + (size_t)bh * HD_ * S_;

    bf16* pr_sh = &pshm[wave][0][0];
    bf16* pi_sh = &pshm[wave][1][0];

    const unsigned smkr_o[2] = { lds_off(smkr[0]), lds_off(smkr[1]) };
    const unsigned smki_o[2] = { lds_off(smki[0]), lds_off(smki[1]) };
    const unsigned svr_o[2]  = { lds_off(svr[0]),  lds_off(svr[1])  };
    const unsigned svi_o[2]  = { lds_off(svi[0]),  lds_off(svi[1])  };

    // Issue one tile's 26 async copies (24 for mk strips, 2 for V^T tile).
    auto stage = [&](int buf, int tbs) {
        const bf16* srcr = mkrb + (size_t)tbs * D_;   // 16 rows x 768, contiguous
        const bf16* srci = mkib + (size_t)tbs * D_;
        #pragma unroll
        for (int j = 0; j < 12; ++j) {                // 1536 16B-chunks / 128 thr
            int c = tid + j * 128;
            async_b128(smkr_o[buf] + c * 16, srcr + c * 8);
        }
        #pragma unroll
        for (int j = 0; j < 12; ++j) {
            int c = tid + j * 128;
            async_b128(smki_o[buf] + c * 16, srci + c * 8);
        }
        int n = tid >> 1, half = tid & 1;             // 64 rows x (2 x 16B) chunks
        unsigned doff = (unsigned)(n * 16 + half * 8) * 2;
        async_b128(svr_o[buf] + doff, vTrb + (size_t)n * S_ + tbs + half * 8);
        async_b128(svi_o[buf] + doff, vTib + (size_t)n * S_ + tbs + half * 8);
    };

    v8f Orr[4], Ori[4], Oir[4], Oii[4];
    #pragma unroll
    for (int g = 0; g < 4; ++g) {
        Orr[g] = zero8(); Ori[g] = zero8();
        Oir[g] = zero8(); Oii[g] = zero8();
    }
    float mR[8], lR[8], mI[8], lI[8];
    #pragma unroll
    for (int r = 0; r < 8; ++r) { mR[r] = -3.0e38f; mI[r] = -3.0e38f; lR[r] = 0.f; lI[r] = 0.f; }

    stage(0, 0);   // prologue: tile 0 into buffer 0

    for (int it = 0; it < S_ / 16; ++it) {
        int tb  = it * 16;
        int cur = it & 1;

        __syncthreads();                 // prior readers of buffer cur^1 done
        if (it + 1 < S_ / 16) {
            stage(cur ^ 1, tb + 16);     // overlap next tile's DMA with compute
            wait_async26();              // current tile complete (in-order retire)
        } else {
            wait_async0();
        }
        __syncthreads();                 // whole block sees staged data

        // ---- scores: sr = qr.mkr^T - qi.mki^T ; si = qr.mki^T + qi.mkr^T ----
        v8f sr = zero8(), si = zero8();
        const bf16* brb = &smkr[cur][0] + l15 * D_ + koff;
        const bf16* bib = &smki[cur][0] + l15 * D_ + koff;
        #pragma unroll 4
        for (int k = 0; k < D_; k += 32) {
            v16bf aR  = ldfrag(qrb + k);
            v16bf aI  = ldfrag(qib + k);
            v16bf bR  = ldfrag(brb + k);
            v16bf bI  = ldfrag(bib + k);
            v16bf aIn = negf(aI);
            sr = WMMA_BF16(aR,  bR, sr);
            sr = WMMA_BF16(aIn, bI, sr);
            si = WMMA_BF16(aR,  bI, si);
            si = WMMA_BF16(aI,  bR, si);
        }

        // bias (per key column) + 1/scale
        float cbR = cbr[(size_t)(b * S_ + tb + l15) * H_ + h] * 0.125f;
        float cbI = cbi[(size_t)(b * S_ + tb + l15) * H_ + h] * 0.125f;

        // ---- dual online softmax (independent real / imag tracks) ----
        #pragma unroll
        for (int r = 0; r < 8; ++r) {
            // real track
            {
                float sv = sr[r] * 0.125f + cbR;
                float mx = sv;
                mx = fmaxf(mx, __shfl_xor(mx, 1, 32));
                mx = fmaxf(mx, __shfl_xor(mx, 2, 32));
                mx = fmaxf(mx, __shfl_xor(mx, 4, 32));
                mx = fmaxf(mx, __shfl_xor(mx, 8, 32));
                float mnew  = fmaxf(mR[r], mx);
                float alpha = __expf(mR[r] - mnew);
                float p     = __expf(sv - mnew);
                float ps = p;
                ps += __shfl_xor(ps, 1, 32);
                ps += __shfl_xor(ps, 2, 32);
                ps += __shfl_xor(ps, 4, 32);
                ps += __shfl_xor(ps, 8, 32);
                lR[r] = lR[r] * alpha + ps;
                mR[r] = mnew;
                #pragma unroll
                for (int g = 0; g < 4; ++g) { Orr[g][r] *= alpha; Ori[g][r] *= alpha; }
                pr_sh[(r + lh * 8) * 16 + l15] = (bf16)p;
            }
            // imag track
            {
                float sv = si[r] * 0.125f + cbI;
                float mx = sv;
                mx = fmaxf(mx, __shfl_xor(mx, 1, 32));
                mx = fmaxf(mx, __shfl_xor(mx, 2, 32));
                mx = fmaxf(mx, __shfl_xor(mx, 4, 32));
                mx = fmaxf(mx, __shfl_xor(mx, 8, 32));
                float mnew  = fmaxf(mI[r], mx);
                float alpha = __expf(mI[r] - mnew);
                float p     = __expf(sv - mnew);
                float ps = p;
                ps += __shfl_xor(ps, 1, 32);
                ps += __shfl_xor(ps, 2, 32);
                ps += __shfl_xor(ps, 4, 32);
                ps += __shfl_xor(ps, 8, 32);
                lI[r] = lI[r] * alpha + ps;
                mI[r] = mnew;
                #pragma unroll
                for (int g = 0; g < 4; ++g) { Oii[g][r] *= alpha; Oir[g][r] *= alpha; }
                pi_sh[(r + lh * 8) * 16 + l15] = (bf16)p;
            }
        }

        // ---- PV: refragment P via LDS, multiply with staged V^T (K=16) ----
        v16bf pR = ldfrag_lo(pr_sh + l15 * 16 + koff);
        v16bf pI = ldfrag_lo(pi_sh + l15 * 16 + koff);
        #pragma unroll
        for (int g = 0; g < 4; ++g) {
            v16bf vR = ldfrag_lo(&svr[cur][0] + (g * 16 + l15) * 16 + koff);
            v16bf vI = ldfrag_lo(&svi[cur][0] + (g * 16 + l15) * 16 + koff);
            Orr[g] = WMMA_BF16(pR, vR, Orr[g]);
            Ori[g] = WMMA_BF16(pR, vI, Ori[g]);
            Oir[g] = WMMA_BF16(pI, vR, Oir[g]);
            Oii[g] = WMMA_BF16(pI, vI, Oii[g]);
        }
    }

    // ---- epilogue: cr = Orr/lR - Oii/lI ; ci = Ori/lR + Oir/lI ----
    const size_t BSD = (size_t)B_ * S_ * D_;
    #pragma unroll
    for (int r = 0; r < 8; ++r) {
        float ilr = 1.0f / lR[r];
        float ili = 1.0f / lI[r];
        int m = r + lh * 8;
        size_t row = (size_t)(b * S_ + srow + m) * D_ + h * HD_;
        #pragma unroll
        for (int g = 0; g < 4; ++g) {
            float cr = Orr[g][r] * ilr - Oii[g][r] * ili;
            float ci = Ori[g][r] * ilr + Oir[g][r] * ili;
            out[row + g * 16 + l15]       = cr;
            out[BSD + row + g * 16 + l15] = ci;
        }
    }
}

// ---------------------------------------------------------------------------
// Host launcher
// ---------------------------------------------------------------------------
extern "C" void kernel_launch(void* const* d_in, const int* in_sizes, int n_in,
                              void* d_out, int out_size, void* d_ws, size_t ws_size,
                              hipStream_t stream)
{
    const float* hidden_r = (const float*)d_in[0];
    const float* hidden_i = (const float*)d_in[1];
    const float* Wq_r = (const float*)d_in[2];
    const float* Wq_i = (const float*)d_in[3];
    const float* Wk_r = (const float*)d_in[4];
    const float* Wk_i = (const float*)d_in[5];
    const float* Wcb_r = (const float*)d_in[6];
    const float* Wcb_i = (const float*)d_in[7];
    const float* Wv_r = (const float*)d_in[8];
    const float* Wv_i = (const float*)d_in[9];
    const float* bv_r = (const float*)d_in[10];
    const float* bv_i = (const float*)d_in[11];
    const float* mix_r = (const float*)d_in[12];
    const float* mix_i = (const float*)d_in[13];

    char* ws = (char*)d_ws;
    size_t off = 0;
    auto alloc = [&](size_t bytes) -> char* {
        char* p = ws + off;
        off += (bytes + 255) & ~(size_t)255;
        return p;
    };

    // bf16 staging buffers (total ~115 MB; fits comfortably in 192 MB L2)
    bf16* hr16 = (bf16*)alloc((size_t)BS_ * D_ * 2);
    bf16* hi16 = (bf16*)alloc((size_t)BS_ * D_ * 2);
    bf16* WqrT = (bf16*)alloc((size_t)D_ * D_ * 2);
    bf16* WqiT = (bf16*)alloc((size_t)D_ * D_ * 2);
    bf16* WkrT = (bf16*)alloc((size_t)D_ * D_ * 2);
    bf16* WkiT = (bf16*)alloc((size_t)D_ * D_ * 2);
    bf16* WvrT = (bf16*)alloc((size_t)D_ * D_ * 2);
    bf16* WviT = (bf16*)alloc((size_t)D_ * D_ * 2);
    bf16* qr16 = (bf16*)alloc((size_t)BS_ * D_ * 2);
    bf16* qi16 = (bf16*)alloc((size_t)BS_ * D_ * 2);
    bf16* kr16 = (bf16*)alloc((size_t)BS_ * D_ * 2);
    bf16* ki16 = (bf16*)alloc((size_t)BS_ * D_ * 2);
    bf16* vr16 = (bf16*)alloc((size_t)BS_ * D_ * 2);
    bf16* vi16 = (bf16*)alloc((size_t)BS_ * D_ * 2);
    bf16* mkrW = (bf16*)alloc((size_t)BH_ * S_ * D_ * 2);
    bf16* mkiW = (bf16*)alloc((size_t)BH_ * S_ * D_ * 2);
    bf16* vTrW = (bf16*)alloc((size_t)BH_ * HD_ * S_ * 2);
    bf16* vTiW = (bf16*)alloc((size_t)BH_ * HD_ * S_ * 2);
    float* cbrW = (float*)alloc((size_t)BS_ * H_ * 4);
    float* cbiW = (float*)alloc((size_t)BS_ * H_ * 4);
    (void)ws_size; (void)in_sizes; (void)n_in; (void)out_size;

    // 1) convert hidden to bf16
    int n1 = BS_ * D_;
    k_cvt<<<(n1 + 255) / 256, 256, 0, stream>>>(hidden_r, hr16, n1);
    k_cvt<<<(n1 + 255) / 256, 256, 0, stream>>>(hidden_i, hi16, n1);

    // 2) transpose + convert weights
    int n2 = D_ * D_;
    int g2 = (n2 + 255) / 256;
    k_cvtT<<<g2, 256, 0, stream>>>(Wq_r, WqrT);
    k_cvtT<<<g2, 256, 0, stream>>>(Wq_i, WqiT);
    k_cvtT<<<g2, 256, 0, stream>>>(Wk_r, WkrT);
    k_cvtT<<<g2, 256, 0, stream>>>(Wk_i, WkiT);
    k_cvtT<<<g2, 256, 0, stream>>>(Wv_r, WvrT);
    k_cvtT<<<g2, 256, 0, stream>>>(Wv_i, WviT);

    // 3) q / k / v complex projections (WMMA)
    k_proj<<<dim3(BS_ / 16, D_ / 64, 3), 128, 0, stream>>>(
        hr16, hi16, WqrT, WqiT, WkrT, WkiT, WvrT, WviT, bv_r, bv_i,
        qr16, qi16, kr16, ki16, vr16, vi16);

    // 4) content bias (tiny)
    k_cb<<<(BS_ * H_ + 255) / 256, 256, 0, stream>>>(
        hidden_r, hidden_i, Wcb_r, Wcb_i, cbrW, cbiW);

    // 5) mixed keys mk = mix ∘ k
    size_t nmk = (size_t)BH_ * S_ * D_;
    k_mk<<<(unsigned)((nmk + 255) / 256), 256, 0, stream>>>(
        kr16, ki16, mix_r, mix_i, mkrW, mkiW);

    // 6) per-head V transpose
    int nvt = BH_ * HD_ * S_;
    k_vT<<<(nvt + 255) / 256, 256, 0, stream>>>(vr16, vi16, vTrW, vTiW);

    // 7) attention (double-buffered async-LDS WMMA flash)
    k_attn<<<dim3(S_ / 64, BH_), 128, 0, stream>>>(
        qr16, qi16, mkrW, mkiW, vTrW, vTiW, cbrW, cbiW, (float*)d_out);
}
